// Decoder_71322226917755
// MI455X (gfx1250) — matrix operand order repaired and checked by reading, not compiled, not run on hardware
//
#include <hip/hip_runtime.h>
#include <hip/hip_bf16.h>
#include <math.h>

typedef __attribute__((ext_vector_type(16))) _Float16 v16h;
typedef __attribute__((ext_vector_type(8)))  _Float16 v8h;
typedef __attribute__((ext_vector_type(8)))  float    v8f;

#define HDIM 512
#define GDIM 1536
#define BSZ  16
#define TD   64
#define TE   64
#define VOC  32000
#define MROWS (BSZ*TD)   // 1024

// ---------------- WMMA fragment loaders (A and B both use contiguous k = 16*half + s) ----

// A (16xK tile, f32 source, convert to f16 in-register)
__device__ inline v16h load_a_f32(const float* A, int lda, int m0, int k0, int lane) {
  int m  = m0 + (lane & 15);
  int hh = (lane >> 4) & 1;                // half-wave selects K sub-run
  const float* p = A + (size_t)m * lda + k0 + hh * 16;
  float4 f0 = *(const float4*)(p);
  float4 f1 = *(const float4*)(p + 4);
  float4 f2 = *(const float4*)(p + 8);
  float4 f3 = *(const float4*)(p + 12);
  v16h a;
  a[0]=(_Float16)f0.x; a[1]=(_Float16)f0.y; a[2]=(_Float16)f0.z; a[3]=(_Float16)f0.w;
  a[4]=(_Float16)f1.x; a[5]=(_Float16)f1.y; a[6]=(_Float16)f1.z; a[7]=(_Float16)f1.w;
  a[8]=(_Float16)f2.x; a[9]=(_Float16)f2.y; a[10]=(_Float16)f2.z; a[11]=(_Float16)f2.w;
  a[12]=(_Float16)f3.x; a[13]=(_Float16)f3.y; a[14]=(_Float16)f3.z; a[15]=(_Float16)f3.w;
  return a;
}

// A (16xK tile, f16 source; works for global or LDS pointers after inlining)
__device__ inline v16h load_a_f16(const _Float16* A, int lda, int m0, int k0, int lane) {
  int m  = m0 + (lane & 15);
  int hh = (lane >> 4) & 1;
  const _Float16* p = A + (size_t)m * lda + k0 + hh * 16;
  v8h r0 = *(const v8h*)(p);
  v8h r1 = *(const v8h*)(p + 8);
  v16h a;
#pragma unroll
  for (int i = 0; i < 8; ++i) { a[i] = r0[i]; a[8 + i] = r1[i]; }
  return a;
}

// B = W^T tile: B[k][n] = W[n0+n][k0+k], W row-major [N][K] f32
__device__ inline v16h load_b_f32(const float* W, int ldw, int n0, int k0, int lane) {
  int n  = n0 + (lane & 15);
  int kb = (lane >> 4) * 16;
  const float* p = W + (size_t)n * ldw + k0 + kb;
  float4 f0 = *(const float4*)(p);
  float4 f1 = *(const float4*)(p + 4);
  float4 f2 = *(const float4*)(p + 8);
  float4 f3 = *(const float4*)(p + 12);
  v16h b;
  b[0]=(_Float16)f0.x; b[1]=(_Float16)f0.y; b[2]=(_Float16)f0.z; b[3]=(_Float16)f0.w;
  b[4]=(_Float16)f1.x; b[5]=(_Float16)f1.y; b[6]=(_Float16)f1.z; b[7]=(_Float16)f1.w;
  b[8]=(_Float16)f2.x; b[9]=(_Float16)f2.y; b[10]=(_Float16)f2.z; b[11]=(_Float16)f2.w;
  b[12]=(_Float16)f3.x; b[13]=(_Float16)f3.y; b[14]=(_Float16)f3.z; b[15]=(_Float16)f3.w;
  return b;
}

__device__ inline v16h load_b_f16(const _Float16* W, int ldw, int n0, int k0, int lane) {
  int n  = n0 + (lane & 15);
  int kb = (lane >> 4) * 16;
  return *(const v16h*)(W + (size_t)n * ldw + k0 + kb);
}

__device__ inline v8f wmma_f16(v16h a, v16h b, v8f c) {
  return __builtin_amdgcn_wmma_f32_16x16x32_f16(false, a, false, b, (short)0, c,
                                                false, false);
}

// ---------------- Generic WMMA GEMM: D[m][n] = sum_k A[m][k]*W[n][k] (+bias) --
// Each wave computes MT vertically-stacked 16x16 tiles per B fragment
// (B loaded once per k-step -> B L2 traffic divided by MT).
// OUT==0: f32 store (+bias). OUT==1: tanh then f16 store (+bias).
template<bool A16, bool B16, int OUT, int MT>
__global__ void gemm_wmma(const void* __restrict__ A_, int lda,
                          const void* __restrict__ W_, int ldw,
                          const float* __restrict__ bias,
                          void* __restrict__ D_, int ldc,
                          int Ntiles, int Ksteps, int total_waves) {
  int wid = (int)((blockIdx.x * blockDim.x + threadIdx.x) >> 5);
  if (wid >= total_waves) return;               // retires whole waves only
  int lane = threadIdx.x & 31;
  int mg = wid / Ntiles, nt = wid - mg * Ntiles;
  int m0 = mg * (MT * 16), n0 = nt << 4;
  v8f c[MT] = {};
  for (int ks = 0; ks < Ksteps; ++ks) {
    int k0 = ks << 5;
    v16h b;
    if constexpr (B16) b = load_b_f16((const _Float16*)W_, ldw, n0, k0, lane);
    else               b = load_b_f32((const float*)W_,    ldw, n0, k0, lane);
#pragma unroll
    for (int i = 0; i < MT; ++i) {
      v16h a;
      if constexpr (A16) a = load_a_f16((const _Float16*)A_, lda, m0 + i * 16, k0, lane);
      else               a = load_a_f32((const float*)A_,    lda, m0 + i * 16, k0, lane);
      c[i] = wmma_f16(a, b, c[i]);
    }
  }
  int col = lane & 15, radd = (lane >> 4) << 3;
  float bv = bias ? bias[n0 + col] : 0.0f;
#pragma unroll
  for (int i = 0; i < MT; ++i) {
    int mrow = m0 + i * 16 + radd;
    if constexpr (OUT == 0) {
      float* D = (float*)D_;
#pragma unroll
      for (int r = 0; r < 8; ++r)
        D[(size_t)(mrow + r) * ldc + n0 + col] = c[i][r] + bv;
    } else {
      _Float16* D = (_Float16*)D_;
#pragma unroll
      for (int r = 0; r < 8; ++r)
        D[(size_t)(mrow + r) * ldc + n0 + col] = (_Float16)tanhf(c[i][r] + bv);
    }
  }
}

// ---------------- Logits GEMM with LDS-staged A panel ------------------------
// Dominant stage: 1024x32000x512. Stage 128 rows of f16 A (128KB LDS) once per
// block; each wave holds 8 accumulators so each W_out fragment is loaded once
// per (N-tile, M-group): W_out L2 traffic = 8 passes instead of 64.
#define LOG_MT   8     // M-tiles per wave (128 rows per block)
#define LOG_NBLK 125   // blocks per M-group
#define LOG_NGRP 2     // N-groups (of 8 N-tiles) per block: 125*2*8 = 2000

__global__ void __launch_bounds__(256)
logits_gemm_k(const _Float16* __restrict__ Ah,   // dense_h 1024x512 f16
              const float* __restrict__ W,       // W_out 32000x512 f32
              const float* __restrict__ bias,    // b_out
              float* __restrict__ D) {           // logits 1024x32000 f32
  __shared__ _Float16 sA[128 * HDIM];            // 128KB (2 blocks / 320KB WGP)
  int mg   = blockIdx.x / LOG_NBLK;              // 0..7
  int nb   = blockIdx.x - mg * LOG_NBLK;         // 0..124
  int tid  = threadIdx.x;
  int lane = tid & 31, wave = tid >> 5;

  // stage A panel: 128 rows x 512 f16 = 8192 x 16B
  {
    const uint4* src = (const uint4*)(Ah + (size_t)mg * 128 * HDIM);
    uint4* dst = (uint4*)sA;
    for (int i = tid; i < 8192; i += 256) dst[i] = src[i];
  }
  __syncthreads();

#pragma unroll
  for (int g = 0; g < LOG_NGRP; ++g) {
    int nt = (nb * LOG_NGRP + g) * 8 + wave;     // 0..1999
    int n0 = nt << 4;
    v8f c[LOG_MT] = {};
    const float* pb = W + (size_t)(n0 + (lane & 15)) * HDIM + ((lane >> 4) * 16);
    for (int ks = 0; ks < 16; ++ks) {
      int k0 = ks << 5;
      __builtin_prefetch(pb + k0 + 32, 0, 1);    // next k-step B line
      v16h b = load_b_f32(W, HDIM, n0, k0, lane);
#pragma unroll
      for (int i = 0; i < LOG_MT; ++i) {
        v16h a = load_a_f16(sA, HDIM, i * 16, k0, lane);   // LDS reads
        c[i] = wmma_f16(a, b, c[i]);
      }
    }
    int col = lane & 15, radd = (lane >> 4) << 3;
    float bv = bias[n0 + col];
#pragma unroll
    for (int i = 0; i < LOG_MT; ++i) {
      int mrow = mg * 128 + i * 16 + radd;
#pragma unroll
      for (int r = 0; r < 8; ++r)
        D[(size_t)(mrow + r) * VOC + n0 + col] = c[i][r] + bv;
    }
  }
}

// ---------------- fused persistent GRU scan ----------------------------------
// 64 sequential steps in ONE kernel: 12 blocks x 256 thr = 96 waves (one per
// hp N-tile). Grid barrier between GEMM phase and gate phase. W_hh f16 stays
// L2-resident; h (32KB) cache-hot. Replaces 128 kernel launches.
#define GRU_BLOCKS 12

__device__ inline void grid_barrier(unsigned* counter, unsigned* gen) {
  __syncthreads();
  if (threadIdx.x == 0) {
    __threadfence();
    unsigned g = __atomic_load_n(gen, __ATOMIC_RELAXED);
    unsigned arrived = atomicAdd(counter, 1u) + 1u;
    if (arrived == (unsigned)GRU_BLOCKS) {
      *counter = 0u;
      __threadfence();
      __atomic_store_n(gen, g + 1u, __ATOMIC_RELEASE);
    } else {
      while (__atomic_load_n(gen, __ATOMIC_ACQUIRE) == g)
        __builtin_amdgcn_s_sleep(1);
    }
  }
  __syncthreads();
  __threadfence();
}

__global__ void __launch_bounds__(256)
gru_scan_k(const float* __restrict__ x_proj, const _Float16* __restrict__ Whh,
           const float* __restrict__ b_hh, float* __restrict__ hbuf,
           float* __restrict__ hp, float* __restrict__ dec_out,
           const int* __restrict__ lengths, unsigned* __restrict__ bar) {
  unsigned* counter = bar;
  unsigned* gen     = bar + 16;     // separate cache lines
  int tid  = threadIdx.x, lane = tid & 31;
  int gtid = blockIdx.x * 256 + tid;
  int w    = gtid >> 5;             // wave 0..95 == hp N-tile index
  int n0   = w << 4;
  int col  = lane & 15, radd = (lane >> 4) << 3;
  float bv = b_hh[n0 + col];

  // h0 = 0
  for (int i = gtid; i < BSZ * HDIM; i += GRU_BLOCKS * 256) hbuf[i] = 0.0f;
  grid_barrier(counter, gen);

  for (int t = 0; t < TD; ++t) {
    // ---- phase 1: hp = h @ W_hh^T + b_hh (one 16x16 tile per wave) ----
    {
      v8f c = {};
      for (int ks = 0; ks < 16; ++ks) {
        int k0 = ks << 5;
        v16h a = load_a_f32(hbuf, HDIM, 0, k0, lane);
        v16h b = load_b_f16(Whh, HDIM, n0, k0, lane);
        c = wmma_f16(a, b, c);
      }
#pragma unroll
      for (int r = 0; r < 8; ++r)
        hp[(size_t)(radd + r) * GDIM + n0 + col] = c[r] + bv;
    }
    grid_barrier(counter, gen);

    // ---- phase 2: gates ----
    for (int idx = gtid; idx < BSZ * HDIM; idx += GRU_BLOCKS * 256) {
      int b = idx >> 9, j = idx & 511;
      const float* xp  = x_proj + (size_t)(b * TD + t) * GDIM;
      const float* hpp = hp + (size_t)b * GDIM;
      float r = 1.0f / (1.0f + __expf(-(xp[j]       + hpp[j])));
      float z = 1.0f / (1.0f + __expf(-(xp[512 + j] + hpp[512 + j])));
      float n = tanhf(xp[1024 + j] + r * hpp[1024 + j]);
      float hprev = hbuf[idx];
      float hnew  = (1.0f - z) * n + z * hprev;
      bool valid  = t < lengths[b];
      hbuf[idx] = valid ? hnew : hprev;
      dec_out[(size_t)(b * TD + t) * HDIM + j] = valid ? hnew : 0.0f;
    }
    grid_barrier(counter, gen);
  }
}

// ---------------- small elementwise kernels ----------------------------------
__global__ void f32_to_f16_k(const float* __restrict__ s, _Float16* __restrict__ d, int n) {
  int i = blockIdx.x * blockDim.x + threadIdx.x;
  if (i < n) d[i] = (_Float16)s[i];
}
__global__ void zero_u32_k(unsigned* p, int n) {
  int i = blockIdx.x * blockDim.x + threadIdx.x;
  if (i < n) p[i] = 0u;
}
__global__ void copy_f32_k(const float* __restrict__ s, float* __restrict__ d, int n) {
  int i = blockIdx.x * blockDim.x + threadIdx.x;
  if (i < n) d[i] = s[i];
}

// energy + masked softmax + context for one (b,d) pair per block (256 thr)
__global__ void attn_k(const float* __restrict__ enc_p, const float* __restrict__ dec_p,
                       const float* __restrict__ enc,
                       const int* __restrict__ enc_len, const int* __restrict__ dec_len,
                       const float* __restrict__ b_attn, const float* __restrict__ v_w,
                       const float* __restrict__ v_b, float* __restrict__ context) {
  int bd = blockIdx.x;            // b*64 + d
  int b = bd >> 6, d = bd & 63;
  __shared__ float sE[TE];
  __shared__ float sA2[TE];
  int tid = threadIdx.x, lane = tid & 31, wave = tid >> 5;   // 8 waves
  bool dvalid = d < dec_len[b];
  int elen = enc_len[b];
  for (int e = wave; e < TE; e += 8) {
    float m = (dvalid && e < elen) ? 1.0f : 0.0f;
    float acc = 0.0f;
    const float* ep = enc_p + (size_t)(b * TE + e) * HDIM;
    const float* dp = dec_p + (size_t)bd * HDIM;
    for (int hx = lane; hx < HDIM; hx += 32)
      acc += v_w[hx] * tanhf(ep[hx] * m + dp[hx] * m + b_attn[hx]);
#pragma unroll
    for (int off = 16; off; off >>= 1) acc += __shfl_xor(acc, off, 32);
    if (lane == 0) sE[e] = acc + v_b[0];
  }
  __syncthreads();
  if (wave == 0) {                 // softmax over 64 entries; 2 per lane
    float e0 = sE[lane], e1 = sE[lane + 32];
    float mx = fmaxf(e0, e1);
#pragma unroll
    for (int off = 16; off; off >>= 1) mx = fmaxf(mx, __shfl_xor(mx, off, 32));
    float x0 = __expf(e0 - mx), x1 = __expf(e1 - mx);
    float sm = x0 + x1;
#pragma unroll
    for (int off = 16; off; off >>= 1) sm += __shfl_xor(sm, off, 32);
    float a0 = x0 / sm, a1 = x1 / sm;
    float m0 = (dvalid && lane < elen) ? 1.0f : 0.0f;
    float m1 = (dvalid && (lane + 32) < elen) ? 1.0f : 0.0f;
    float am0 = a0 * m0, am1 = a1 * m1;
    float s = am0 + am1;
#pragma unroll
    for (int off = 16; off; off >>= 1) s += __shfl_xor(s, off, 32);
    s = (s == 0.0f) ? 1.0f : s;
    sA2[lane] = am0 / s; sA2[lane + 32] = am1 / s;
  }
  __syncthreads();
  for (int hx = tid; hx < HDIM; hx += 256) {
    float acc = 0.0f;
#pragma unroll 4
    for (int e = 0; e < TE; ++e)
      acc += sA2[e] * enc[(size_t)(b * TE + e) * HDIM + hx];
    context[(size_t)bd * HDIM + hx] = acc;
  }
}

__global__ void concat_k(const float* __restrict__ dec_out, const float* __restrict__ ctx,
                         float* __restrict__ cc) {
  int idx = blockIdx.x * blockDim.x + threadIdx.x;
  if (idx >= MROWS * 2 * HDIM) return;
  int row = idx >> 10, k = idx & 1023;
  cc[idx] = (k < HDIM) ? dec_out[(size_t)row * HDIM + k]
                       : ctx[(size_t)row * HDIM + (k - HDIM)];
}

// ---------------- host side --------------------------------------------------
extern "C" void kernel_launch(void* const* d_in, const int* in_sizes, int n_in,
                              void* d_out, int out_size, void* d_ws, size_t ws_size,
                              hipStream_t stream) {
  const float* input_seqs = (const float*)d_in[0];
  const int*   input_len  = (const int*)  d_in[1];
  const float* enc        = (const float*)d_in[2];
  const int*   enc_len    = (const int*)  d_in[3];
  const float* W_ih       = (const float*)d_in[4];
  const float* W_hh       = (const float*)d_in[5];
  const float* b_ih       = (const float*)d_in[6];
  const float* b_hh       = (const float*)d_in[7];
  const float* W_attn     = (const float*)d_in[8];
  const float* b_attn     = (const float*)d_in[9];
  const float* v_w        = (const float*)d_in[10];
  const float* v_b        = (const float*)d_in[11];
  const float* W_dense    = (const float*)d_in[12];
  const float* b_dense    = (const float*)d_in[13];
  const float* W_out      = (const float*)d_in[14];
  const float* b_out      = (const float*)d_in[15];
  float* out = (float*)d_out;

  char* ws = (char*)d_ws;
  size_t off = 0;
  auto alloc = [&](size_t bytes) -> char* {
    char* p = ws + off; off += (bytes + 255) & ~(size_t)255; return p;
  };
  _Float16* W_hh_h  = (_Float16*)alloc((size_t)GDIM * HDIM * 2);
  float*    x_proj  = (float*)   alloc((size_t)MROWS * GDIM * 4);
  float*    hp      = (float*)   alloc((size_t)BSZ * GDIM * 4);
  float*    hbuf    = (float*)   alloc((size_t)BSZ * HDIM * 4);
  float*    dec_out = (float*)   alloc((size_t)MROWS * HDIM * 4);
  float*    enc_p   = (float*)   alloc((size_t)MROWS * HDIM * 4);
  float*    dec_p   = (float*)   alloc((size_t)MROWS * HDIM * 4);
  float*    ctx     = (float*)   alloc((size_t)MROWS * HDIM * 4);
  float*    cc      = (float*)   alloc((size_t)MROWS * 2 * HDIM * 4);
  _Float16* dense_h = (_Float16*)alloc((size_t)MROWS * HDIM * 2);
  unsigned* bar     = (unsigned*)alloc(256);

  // 1) W_hh -> f16 (reused 64x; L2-resident); reset grid-barrier state
  f32_to_f16_k<<<(GDIM * HDIM + 255) / 256, 256, 0, stream>>>(W_hh, W_hh_h, GDIM * HDIM);
  zero_u32_k<<<1, 64, 0, stream>>>(bar, 64);

  // 2) x_proj = X @ W_ih^T + b_ih : 1024x1536, K=512; MT=4 -> 16x96=1536 waves
  gemm_wmma<false, false, 0, 4><<<1536 / 8, 256, 0, stream>>>(
      input_seqs, HDIM, W_ih, HDIM, b_ih, x_proj, GDIM, 96, 16, 1536);

  // 3) fused 64-step GRU scan (single persistent kernel, grid barriers)
  gru_scan_k<<<GRU_BLOCKS, 256, 0, stream>>>(
      x_proj, W_hh_h, b_hh, hbuf, hp, dec_out, input_len, bar);

  // 4) attention projections: enc_p = enc @ We^T, dec_p = dec @ Wd^T (MT=4)
  gemm_wmma<false, false, 0, 4><<<512 / 8, 256, 0, stream>>>(
      enc, HDIM, W_attn, 2 * HDIM, nullptr, enc_p, HDIM, 32, 16, 512);
  gemm_wmma<false, false, 0, 4><<<512 / 8, 256, 0, stream>>>(
      dec_out, HDIM, W_attn + HDIM, 2 * HDIM, nullptr, dec_p, HDIM, 32, 16, 512);

  // 5) energy + masked softmax + context, one block per (b,d)
  attn_k<<<BSZ * TD, 256, 0, stream>>>(enc_p, dec_p, enc, enc_len, input_len,
                                       b_attn, v_w, v_b, ctx);

  // 6) concat(dec_out, ctx) -> dense = tanh(cc @ W_dense^T + b), store f16
  concat_k<<<(MROWS * 2 * HDIM + 255) / 256, 256, 0, stream>>>(dec_out, ctx, cc);
  gemm_wmma<false, false, 1, 4><<<512 / 8, 256, 0, stream>>>(
      cc, 2 * HDIM, W_dense, 2 * HDIM, b_dense, dense_h, HDIM, 32, 32, 512);

  // 7) logits = dense @ W_out^T + b_out (dominant stage), LDS-staged A panel
  logits_gemm_k<<<8 * LOG_NBLK, 256, 0, stream>>>(dense_h, W_out, b_out, out);

  // 8) h_last appended after logits
  copy_f32_k<<<(BSZ * HDIM + 255) / 256, 256, 0, stream>>>(
      hbuf, out + (size_t)MROWS * VOC, BSZ * HDIM);
}
